// LlamaAttention_34892314313606
// MI455X (gfx1250) — matrix operand order repaired and checked by reading, not compiled, not run on hardware
//
#include <hip/hip_runtime.h>
#include <hip/hip_bf16.h>
#include <stdint.h>

typedef __attribute__((ext_vector_type(16))) __bf16 v16bf;
typedef __attribute__((ext_vector_type(8)))  float  v8f;
typedef __attribute__((ext_vector_type(4)))  unsigned int u32x4;
typedef __attribute__((ext_vector_type(8)))  int  i32x8;
typedef __attribute__((ext_vector_type(4)))  int  i32x4;

#define DEVFN __device__ __forceinline__

#ifndef __has_builtin
#define __has_builtin(x) 0
#endif
#if __has_builtin(__builtin_amdgcn_tensor_load_to_lds) && \
    __has_builtin(__builtin_amdgcn_s_wait_tensorcnt)
#define USE_TDM 1
#else
#define USE_TDM 0
#endif

constexpr int Bd  = 2;
constexpr int Sd  = 2048;
constexpr int HIDc = 2048;
constexpr int NHc  = 16;
constexpr int HDc  = 128;
constexpr int Mdim = Bd * Sd;            // 4096
constexpr float SCALE = 0.08838834764831845f; // 1/sqrt(128)

DEVFN unsigned short f2bf(float f) {
  unsigned int u = __float_as_uint(f);
  u += 0x7fffu + ((u >> 16) & 1u);      // round-to-nearest-even
  return (unsigned short)(u >> 16);
}
DEVFN float bf2f(unsigned short h) {
  return __uint_as_float(((unsigned int)h) << 16);
}

union FragAB { v16bf v; unsigned int u[8]; };

#if USE_TDM
// 2-D Tensor-Data-Mover load: global (row-major, bf16) -> padded LDS tile.
// Descriptor packing per CDNA5 ISA (D# groups 0/1).
// 6-arg builtin form: (g0, g1, g2, g3, extra, cpol).
DEVFN void tdm_load_2d(unsigned lds_addr, const void* gptr,
                       unsigned tensor_d0, unsigned tensor_d1,
                       unsigned tile_d0, unsigned tile_d1,
                       unsigned stride_d0,
                       unsigned pad_interval, unsigned pad_amount) {
  unsigned long long ga = (unsigned long long)gptr;
  u32x4 g0;
  g0[0] = 1u;                                   // count=1, user descriptor
  g0[1] = lds_addr;                             // LDS byte address
  g0[2] = (unsigned)ga;                         // global_addr[31:0]
  g0[3] = (unsigned)((ga >> 32) & 0x01ffffffu)  // global_addr[56:32]
          | (2u << 30);                         // type=2 ("image")
  i32x8 g1;
  g1[0] = (int)((1u << 16)                      // data_size = 2 bytes
              | (1u << 20)                      // pad_enable
              | (pad_interval << 22)
              | (pad_amount << 25));
  g1[1] = (int)((tensor_d0 & 0xffffu) << 16);   // tensor_dim0[15:0]
  g1[2] = (int)((tensor_d0 >> 16) | ((tensor_d1 & 0xffffu) << 16));
  g1[3] = (int)((tensor_d1 >> 16) | (tile_d0 << 16));
  g1[4] = (int)(tile_d1 & 0xffffu);             // tile_dim1 (tile_dim2=0)
  g1[5] = (int)stride_d0;                       // tensor_dim0_stride[31:0]
  g1[6] = 0;
  g1[7] = 0;
  i32x4 z4 = {0, 0, 0, 0};                      // 2-D: groups 2/3 disabled
  i32x8 z8 = {0, 0, 0, 0, 0, 0, 0, 0};
  __builtin_amdgcn_tensor_load_to_lds(g0, g1, z4, z4, z8, 0);
}
#endif

// ---------------------------------------------------------------- converts
__global__ void k_f32_to_bf16(const float* __restrict__ src,
                              unsigned short* __restrict__ dst, int n) {
  int i = blockIdx.x * blockDim.x + threadIdx.x;
  int stride = gridDim.x * blockDim.x;
  for (; i < n; i += stride) dst[i] = f2bf(src[i]);
}

// ---------------------------------------------------------------- NT GEMM
// C[M,N] = A[M,K] * B[N,K]^T + bias   (A,B bf16, fp32 accum)
// 128x128 block tile, 8 waves of 64x32, double-buffered LDS filled by TDM.
// MODE 0: fp32 row-major out; MODE 1: bf16 out in (B, NH, S, HD) layout.
template<int MODE>
__global__ __launch_bounds__(256)
void k_gemm_nt(const unsigned short* __restrict__ A,
               const unsigned short* __restrict__ Bw,
               const float* __restrict__ bias,
               void* __restrict__ outp) {
  constexpr int K = HIDc, N = HIDc;
  constexpr int NCH = K / 32;
  __shared__ __align__(16) unsigned short As[2][128][40];
  __shared__ __align__(16) unsigned short Bs[2][128][40];
  const int tid = threadIdx.x;
  const int lane = tid & 31, wid = tid >> 5;
  const int m0 = blockIdx.y * 128, n0 = blockIdx.x * 128;
  const int wm = (wid >> 2) * 64, wn = (wid & 3) * 32;
  const int l15 = lane & 15, h2 = lane >> 4;

  auto fill = [&](int buf, int ci) {
    const int kc = ci * 32;
#if USE_TDM
    if (wid == 0) {
      // 32 bf16 per row = 16 dwords data + 4 dwords pad -> interval=3, amount=3
      tdm_load_2d((unsigned)(size_t)&As[buf][0][0], &A[(size_t)m0 * K + kc],
                  K, Mdim, 32, 128, K, 3, 3);
      tdm_load_2d((unsigned)(size_t)&Bs[buf][0][0], &Bw[(size_t)n0 * K + kc],
                  K, N, 32, 128, K, 3, 3);
    }
#else
    for (int i = tid; i < 1024; i += 256) {
      int r = i >> 3, c = (i & 7) * 4;
      *(uint2*)&As[buf][r][c] = *(const uint2*)&A[(size_t)(m0 + r) * K + kc + c];
      *(uint2*)&Bs[buf][r][c] = *(const uint2*)&Bw[(size_t)(n0 + r) * K + kc + c];
    }
#endif
  };

  v8f acc[4][2] = {};
  fill(0, 0);
  for (int ci = 0; ci < NCH; ++ci) {
#if USE_TDM
    if (wid == 0) __builtin_amdgcn_s_wait_tensorcnt(0);
#endif
    __syncthreads();
    if (ci + 1 < NCH) fill((ci + 1) & 1, ci + 1);
    const int buf = ci & 1;
    FragAB a[4], b[2];
    for (int i = 0; i < 4; ++i) {
      const int row = wm + i * 16 + l15;
      for (int v = 0; v < 8; ++v) {
        const int ka = ((v & 4) << 2) + h2 * 8 + ((v & 3) << 1);
        a[i].u[v] = *(const unsigned int*)&As[buf][row][ka];
      }
    }
    for (int j = 0; j < 2; ++j) {
      const int col = wn + j * 16 + l15;
      for (int v = 0; v < 8; ++v)
        b[j].u[v] = *(const unsigned int*)&Bs[buf][col][((v & 4) << 2) + h2 * 8 + ((v & 3) << 1)];
    }
    for (int i = 0; i < 4; ++i)
      for (int j = 0; j < 2; ++j)
        acc[i][j] = __builtin_amdgcn_wmma_f32_16x16x32_bf16(
            false, a[i].v, false, b[j].v, (short)0, acc[i][j], false, false);
  }
  for (int i = 0; i < 4; ++i)
    for (int j = 0; j < 2; ++j)
      for (int v = 0; v < 8; ++v) {
        const int gm = m0 + wm + i * 16 + h2 * 8 + v;
        const int gn = n0 + wn + j * 16 + l15;
        const float val = acc[i][j][v] + bias[gn];
        if (MODE == 0) {
          ((float*)outp)[(size_t)gm * N + gn] = val;
        } else {
          const int b_ = gm >> 11, s_ = gm & (Sd - 1);
          const int h_ = gn >> 7,  d_ = gn & (HDc - 1);
          ((unsigned short*)outp)[((size_t)(b_ * NHc + h_) * Sd + s_) * HDc + d_] =
              f2bf(val);
        }
      }
}

// ---------------------------------------------------------------- RoPE
__global__ void k_rope(unsigned short* __restrict__ q,
                       unsigned short* __restrict__ k,
                       const int* __restrict__ pos_ids) {
  const int idx = blockIdx.x * blockDim.x + threadIdx.x;
  constexpr int total = Bd * NHc * Sd * (HDc / 2);
  if (idx >= total) return;
  const int i = idx & 63;
  const int s = (idx >> 6) & (Sd - 1);
  const int h = (idx >> 17) & (NHc - 1);
  const int b = idx >> 21;
  const int pos = pos_ids[b * Sd + s];
  const float freq = __powf(10000.0f, -(float)i * (1.0f / 64.0f));
  const float ang = (float)pos * freq;
  const float c = __cosf(ang), sn = __sinf(ang);
  const size_t base = ((size_t)(b * NHc + h) * Sd + s) * HDc + i;
  {
    float x1 = bf2f(q[base]), x2 = bf2f(q[base + 64]);
    q[base]      = f2bf(x1 * c - x2 * sn);
    q[base + 64] = f2bf(x2 * c + x1 * sn);
  }
  {
    float x1 = bf2f(k[base]), x2 = bf2f(k[base + 64]);
    k[base]      = f2bf(x1 * c - x2 * sn);
    k[base + 64] = f2bf(x2 * c + x1 * sn);
  }
}

// ---------------------------------------------------------------- flash attn
__global__ __launch_bounds__(128)
void k_flash(const unsigned short* __restrict__ qb,
             const unsigned short* __restrict__ kb,
             const unsigned short* __restrict__ vb,
             unsigned short* __restrict__ attnb) {
  __shared__ __align__(16) unsigned short Ks[32][132];  // K tile (key, d), padded
  __shared__ __align__(16) unsigned short Vt[128][36];  // V tile transposed (d, key)
  __shared__ __align__(16) unsigned short Pb[4][16][40];// per-wave P relayout
  const int tid = threadIdx.x, lane = tid & 31, wid = tid >> 5;
  const int l15 = lane & 15, h2 = lane >> 4;
  const int qt = blockIdx.x, h = blockIdx.y, b = blockIdx.z;
  const size_t bh = (size_t)(b * NHc + h) * Sd * HDc;
  const int qbase = qt * 64 + wid * 16;

  // Q fragments for this wave's 16 rows (A-layout, 4 chunks of K=32 in d)
  FragAB qa[4];
  {
    const unsigned short* qrow = qb + bh + (size_t)(qbase + l15) * HDc;
    for (int c = 0; c < 4; ++c)
      for (int v = 0; v < 8; ++v) {
        const int d = c * 32 + ((v & 4) << 2) + h2 * 8 + ((v & 3) << 1);
        qa[c].u[v] = *(const unsigned int*)&qrow[d];
      }
  }

  v8f o[8] = {};
  float mrun[8], lrun[8];
  for (int v = 0; v < 8; ++v) { mrun[v] = -1e30f; lrun[v] = 0.f; }

  const int ntiles = qt * 2 + 2;                 // causal: keys <= qbase_block+63
  for (int t = 0; t < ntiles; ++t) {
    const int k0 = t * 32;
    __syncthreads();
#if USE_TDM
    if (wid == 0)
      // 128 bf16 per row = 64 dwords data + 2 dwords pad -> interval=5, amount=1
      tdm_load_2d((unsigned)(size_t)&Ks[0][0], &kb[bh + (size_t)k0 * HDc],
                  HDc, Sd, 128, 32, HDc, 5, 1);
#else
    for (int i = tid; i < 1024; i += 128) {
      const int r = i >> 5, c4 = (i & 31) * 4;
      *(uint2*)&Ks[r][c4] = *(const uint2*)&kb[bh + (size_t)(k0 + r) * HDc + c4];
    }
#endif
    for (int i = tid; i < 1024; i += 128) {      // V^T staging (transpose)
      const int r = i >> 5, c4 = (i & 31) * 4;
      uint2 w = *(const uint2*)&vb[bh + (size_t)(k0 + r) * HDc + c4];
      Vt[c4 + 0][r] = (unsigned short)(w.x & 0xffff);
      Vt[c4 + 1][r] = (unsigned short)(w.x >> 16);
      Vt[c4 + 2][r] = (unsigned short)(w.y & 0xffff);
      Vt[c4 + 3][r] = (unsigned short)(w.y >> 16);
    }
    if (t + 1 < ntiles) {                        // prefetch next tiles into L2
      __builtin_prefetch(&kb[bh + (size_t)(k0 + 32) * HDc + tid * 32], 0, 1);
      __builtin_prefetch(&vb[bh + (size_t)(k0 + 32) * HDc + tid * 32], 0, 1);
    }
#if USE_TDM
    if (wid == 0) __builtin_amdgcn_s_wait_tensorcnt(0);
#endif
    __syncthreads();

    // S = Q K^T  (two 16-key fragments), then scale + causal mask
    float sc[2][8];
    for (int j = 0; j < 2; ++j) {
      v8f s = {};
      const int keyl = j * 16 + l15;
      for (int c = 0; c < 4; ++c) {
        FragAB kf;
        for (int v = 0; v < 8; ++v)
          kf.u[v] = *(const unsigned int*)&Ks[keyl][c * 32 + ((v & 4) << 2) + h2 * 8 + ((v & 3) << 1)];
        s = __builtin_amdgcn_wmma_f32_16x16x32_bf16(
            false, qa[c].v, false, kf.v, (short)0, s, false, false);
      }
      const int key = k0 + keyl;
      for (int v = 0; v < 8; ++v) {
        const int row = qbase + h2 * 8 + v;
        sc[j][v] = s[v] * SCALE + (key <= row ? 0.f : -1e30f);
      }
    }

    // online softmax (row stats replicated across each 16-lane half)
    float alpha[8], p[2][8];
    for (int v = 0; v < 8; ++v) {
      float r = fmaxf(sc[0][v], sc[1][v]);
      for (int off = 1; off < 16; off <<= 1) r = fmaxf(r, __shfl_xor(r, off, 32));
      const float mn = fmaxf(mrun[v], r);
      alpha[v] = __expf(mrun[v] - mn);
      mrun[v] = mn;
      const float p0 = __expf(sc[0][v] - mn);
      const float p1 = __expf(sc[1][v] - mn);
      p[0][v] = p0; p[1][v] = p1;
      float rs = p0 + p1;
      for (int off = 1; off < 16; off <<= 1) rs += __shfl_xor(rs, off, 32);
      lrun[v] = lrun[v] * alpha[v] + rs;
    }

    // C-fragment -> A-fragment relayout of P through per-wave LDS (in-order DS)
    for (int j = 0; j < 2; ++j)
      for (int v = 0; v < 8; ++v)
        Pb[wid][h2 * 8 + v][j * 16 + l15] = f2bf(p[j][v]);
    FragAB pa;
    for (int v = 0; v < 8; ++v) {
      const int kk = ((v & 4) << 2) + h2 * 8 + ((v & 3) << 1);
      pa.u[v] = *(const unsigned int*)&Pb[wid][l15][kk];
    }

    // rescale O and accumulate P·V over 8 d-chunks of 16
    for (int c = 0; c < 8; ++c) {
      for (int v = 0; v < 8; ++v) o[c][v] *= alpha[v];
      FragAB vf;
      for (int v = 0; v < 8; ++v)
        vf.u[v] = *(const unsigned int*)&Vt[c * 16 + l15][((v & 4) << 2) + h2 * 8 + ((v & 3) << 1)];
      o[c] = __builtin_amdgcn_wmma_f32_16x16x32_bf16(
          false, pa.v, false, vf.v, (short)0, o[c], false, false);
    }
  }

  // normalize and write to attn buffer (B, S, HID) bf16
  for (int c = 0; c < 8; ++c)
    for (int v = 0; v < 8; ++v) {
      const int srow = qbase + h2 * 8 + v;
      const int col = h * HDc + c * 16 + l15;
      attnb[(size_t)(b * Sd + srow) * HIDc + col] = f2bf(o[c][v] / lrun[v]);
    }
}

// ---------------------------------------------------------------- launcher
extern "C" void kernel_launch(void* const* d_in, const int* in_sizes, int n_in,
                              void* d_out, int out_size, void* d_ws, size_t ws_size,
                              hipStream_t stream) {
  (void)in_sizes; (void)n_in; (void)out_size; (void)ws_size;
  const float* hs = (const float*)d_in[0];
  const float* wq = (const float*)d_in[1];
  const float* bq = (const float*)d_in[2];
  const float* wk = (const float*)d_in[3];
  const float* bk = (const float*)d_in[4];
  const float* wv = (const float*)d_in[5];
  const float* bv = (const float*)d_in[6];
  const float* wo = (const float*)d_in[7];
  const float* bo = (const float*)d_in[8];
  const int*  pos = (const int*)d_in[10];

  char* ws = (char*)d_ws;
  size_t off = 0;
  auto carve = [&](size_t elems) {
    unsigned short* p = (unsigned short*)(ws + off);
    off = (off + elems * 2 + 255) & ~(size_t)255;
    return p;
  };
  const size_t nHS = (size_t)Mdim * HIDc;
  const size_t nW  = (size_t)HIDc * HIDc;
  unsigned short* hsb = carve(nHS);
  unsigned short* wqb = carve(nW);
  unsigned short* wkb = carve(nW);
  unsigned short* wvb = carve(nW);
  unsigned short* wob = carve(nW);
  unsigned short* qbuf = carve(nHS);
  unsigned short* kbuf = carve(nHS);
  unsigned short* vbuf = carve(nHS);
  unsigned short* attnb = carve(nHS);

  k_f32_to_bf16<<<(int)((nHS + 255) / 256), 256, 0, stream>>>(hs, hsb, (int)nHS);
  k_f32_to_bf16<<<(int)((nW + 255) / 256), 256, 0, stream>>>(wq, wqb, (int)nW);
  k_f32_to_bf16<<<(int)((nW + 255) / 256), 256, 0, stream>>>(wk, wkb, (int)nW);
  k_f32_to_bf16<<<(int)((nW + 255) / 256), 256, 0, stream>>>(wv, wvb, (int)nW);
  k_f32_to_bf16<<<(int)((nW + 255) / 256), 256, 0, stream>>>(wo, wob, (int)nW);

  const dim3 gg(HIDc / 128, Mdim / 128);
  k_gemm_nt<1><<<gg, 256, 0, stream>>>(hsb, wqb, bq, qbuf);
  k_gemm_nt<1><<<gg, 256, 0, stream>>>(hsb, wkb, bk, kbuf);
  k_gemm_nt<1><<<gg, 256, 0, stream>>>(hsb, wvb, bv, vbuf);

  constexpr int nRope = Bd * NHc * Sd * (HDc / 2);
  k_rope<<<(nRope + 255) / 256, 256, 0, stream>>>(qbuf, kbuf, pos);

  k_flash<<<dim3(Sd / 64, NHc, Bd), 128, 0, stream>>>(qbuf, kbuf, vbuf, attnb);

  k_gemm_nt<0><<<gg, 256, 0, stream>>>(attnb, wob, bo, (float*)d_out);
}
